// Attention_27599459844832
// MI455X (gfx1250) — compile-verified, matrix operand order checked by reading, not run on hardware
//
#include <hip/hip_runtime.h>
#include <hip/hip_bf16.h>
#include <math.h>

// Shapes from the reference
#define Bq 32
#define Sq 1024
#define Hq 1024

typedef __attribute__((ext_vector_type(16))) __bf16 v16bf;
typedef __attribute__((ext_vector_type(8)))  float  v8f;

__device__ __forceinline__ unsigned short f32_to_bf16(float f) {
  unsigned int u = __float_as_uint(f);
  u += 0x7FFFu + ((u >> 16) & 1u);   // round-to-nearest-even
  return (unsigned short)(u >> 16);
}

// Branch-free tanh via hardware exp2/rcp (v_exp_f32 + v_rcp_f32).
// tanh(x) = 1 - 2/(1 + exp(2x)); exp2 overflow/underflow saturates to +/-1.
#define TWO_LOG2E 2.8853900817779268f   // 2*log2(e)
__device__ __forceinline__ float fast_tanh(float x) {
  float e = __builtin_amdgcn_exp2f(x * TWO_LOG2E);
  return 1.0f - 2.0f * __builtin_amdgcn_rcpf(e + 1.0f);
}
#define LOG2E 1.4426950408889634f
__device__ __forceinline__ float fast_exp(float x) {
  return __builtin_amdgcn_exp2f(x * LOG2E);
}

// ---------------------------------------------------------------------------
// Pass 1: WkT[n][k] = bf16(Wk[k][n]).  32x32 LDS tile transpose, both global
// accesses coalesced.  WkT (2 MB) stays resident in the 192 MB L2.
// ---------------------------------------------------------------------------
__global__ __launch_bounds__(256) void wk_transpose_bf16(
    const float* __restrict__ Wk, unsigned short* __restrict__ wkT) {
  __shared__ float t[32][33];
  const int tx = threadIdx.x, ty = threadIdx.y;
  const int nblk = blockIdx.x * 32, kblk = blockIdx.y * 32;
#pragma unroll
  for (int j = 0; j < 4; ++j) {
    int k = kblk + ty + j * 8;
    t[ty + j * 8][tx] = Wk[(size_t)k * Hq + nblk + tx];
  }
  __syncthreads();
#pragma unroll
  for (int j = 0; j < 4; ++j) {
    int n = nblk + ty + j * 8;
    wkT[(size_t)n * Hq + kblk + tx] = f32_to_bf16(t[tx][ty + j * 8]);
  }
}

// ---------------------------------------------------------------------------
// Pass 2: query[b][h] = sum_k dh[b][k] * Wq[k][h].  0.067 GFLOP: plain fp32,
// Wq reads coalesced, dh reads are block-uniform (scalarized).
// ---------------------------------------------------------------------------
__global__ __launch_bounds__(256) void query_gemm(
    const float* __restrict__ dh, const float* __restrict__ Wq,
    float* __restrict__ query) {
  const int idx = blockIdx.x * 256 + threadIdx.x;   // b*H + h
  const int b = idx >> 10, ho = idx & (Hq - 1);
  const float* dhr = dh + b * Hq;
  float acc = 0.f;
#pragma unroll 8
  for (int k = 0; k < Hq; ++k) acc = fmaf(dhr[k], Wq[(size_t)k * Hq + ho], acc);
  query[idx] = acc;
}

// ---------------------------------------------------------------------------
// Pass 3 (dominant, 68.7 GFLOP): fused keys-GEMM + tanh + dot(We) -> energy.
// Grid 512 x 256 threads (8 waves).  Each WG stages 64 encoder rows (bf16,
// row stride padded to 1040 halves to spread LDS banks) and never writes
// "keys" to memory.  Waves 0-3 / 4-7 split the N range of row-tiles 0-3.
// Inner loop: 1024 v_wmma_f32_16x16x32_bf16 per wave; the tanh/We epilogue
// is branch-free VALU+TRANS, which co-executes with the WMMA pipe.
// ---------------------------------------------------------------------------
__global__ __launch_bounds__(256) void keys_energy_wmma(
    const float* __restrict__ enc, const unsigned short* __restrict__ wkT,
    const float* __restrict__ query, const float* __restrict__ We,
    float* __restrict__ energy) {
  extern __shared__ __align__(32) unsigned short Atile[];   // 64 x 1040 bf16
  __shared__ float ered[8][16];
  const int tid = threadIdx.x;

  // Stage 64 rows x 1024 cols of encoder outputs into LDS as bf16.
  const size_t rowBase = (size_t)blockIdx.x * 64;           // flattened b*S+s
  const float4* encv = (const float4*)(enc + rowBase * Hq);
#pragma unroll 4
  for (int it = 0; it < 64; ++it) {
    int idx = it * 256 + tid;            // float4 index, 0..16383
    int row = idx >> 8;                  // 256 float4 per row
    int colv = (idx & 255) * 4;
    float4 v = encv[idx];
    unsigned short* p = &Atile[row * 1040 + colv];
    p[0] = f32_to_bf16(v.x); p[1] = f32_to_bf16(v.y);
    p[2] = f32_to_bf16(v.z); p[3] = f32_to_bf16(v.w);
  }
  __syncthreads();

  const int w = tid >> 5, lane = tid & 31;
  const int tile = w & 3;                 // which 16-row tile
  const int nbBase = (w >> 2) * 32;       // which half of the 64 N-chunks
  const int rowTile = (int)rowBase + tile * 16;
  const int b = rowTile >> 10;            // 64 | 1024 -> batch uniform per WG
  const int lrow = lane & 15;
  const int khalf = (lane < 16) ? 0 : 16; // contiguous K split (k = 16h + s), this code's order
  const int mbase = (lane < 16) ? 0 : 8;  // f32 C/D layout M split
  const unsigned short* ap0 = &Atile[(tile * 16 + lrow) * 1040 + khalf];

  float ep[8] = {0.f, 0.f, 0.f, 0.f, 0.f, 0.f, 0.f, 0.f};

  for (int nb = nbBase; nb < nbBase + 32; ++nb) {
    const int n = nb * 16 + lrow;                       // this lane's column
    const unsigned short* bp0 = wkT + (size_t)n * Hq + khalf;
    v8f c = {};
#pragma unroll 8
    for (int kb = 0; kb < 32; ++kb) {                   // K = 1024 in steps of 32
      v16bf a  = *(const v16bf*)(ap0 + kb * 32);        // 2x ds_load_b128
      v16bf bb = *(const v16bf*)(bp0 + kb * 32);        // 2x global_load_b128 (L2)
      c = __builtin_amdgcn_wmma_f32_16x16x32_bf16(
              false, a, false, bb, (short)0, c, false, false);
    }
    // Fused epilogue: energy += tanh(query + keys) * We  over this N chunk.
    const float qv  = query[b * Hq + n];
    const float wev = We[n];
#pragma unroll
    for (int r = 0; r < 8; ++r) ep[r] += fast_tanh(qv + c[r]) * wev;
  }

  // Reduce across the 16 lanes of each half (N dimension of the C layout).
#pragma unroll
  for (int r = 0; r < 8; ++r) {
    float v = ep[r];
    v += __shfl_xor(v, 1);  v += __shfl_xor(v, 2);
    v += __shfl_xor(v, 4);  v += __shfl_xor(v, 8);
    ep[r] = v;
  }
  if ((lane & 15) == 0) {
#pragma unroll
    for (int r = 0; r < 8; ++r) ered[w][mbase + r] = ep[r];
  }
  __syncthreads();
  // Combine the two N-halves (waves w and w+4) and write 64 energies.
  if (tid < 64) {
    int tt = tid >> 4, rr = tid & 15;
    energy[rowBase + tid] = ered[tt][rr] + ered[tt + 4][rr];
  }
}

// ---------------------------------------------------------------------------
// Pass 4: softmax over S per batch.  One WG per batch row.
// ---------------------------------------------------------------------------
__global__ __launch_bounds__(256) void softmax_s(
    const float* __restrict__ energy, float* __restrict__ attn) {
  __shared__ float red[256];
  const int b = blockIdx.x, tid = threadIdx.x;
  const float* e = energy + b * Sq;
  float ev[4];
  float mx = -INFINITY;
#pragma unroll
  for (int i = 0; i < 4; ++i) { ev[i] = e[tid + i * 256]; mx = fmaxf(mx, ev[i]); }
  red[tid] = mx; __syncthreads();
  for (int off = 128; off > 0; off >>= 1) {
    if (tid < off) red[tid] = fmaxf(red[tid], red[tid + off]);
    __syncthreads();
  }
  mx = red[0]; __syncthreads();
  float s = 0.f;
#pragma unroll
  for (int i = 0; i < 4; ++i) { ev[i] = fast_exp(ev[i] - mx); s += ev[i]; }
  red[tid] = s; __syncthreads();
  for (int off = 128; off > 0; off >>= 1) {
    if (tid < off) red[tid] += red[tid + off];
    __syncthreads();
  }
  const float inv = 1.0f / red[0];
  float* o = attn + b * Sq;
#pragma unroll
  for (int i = 0; i < 4; ++i) o[tid + i * 256] = ev[i] * inv;
}

// ---------------------------------------------------------------------------
// Pass 5: context[b][h] = sum_s attn[b][s] * enc[b][s][h].  Memory-bound
// (134 MB, partially L2-resident after pass 3).  attn staged in LDS once.
// ---------------------------------------------------------------------------
__global__ __launch_bounds__(256) void context_gemv(
    const float* __restrict__ enc, const float* __restrict__ attn,
    float* __restrict__ ctx) {
  __shared__ float aw[Sq];
  const int tid = threadIdx.x;
  const int b = blockIdx.y;
  const int h = blockIdx.x * 256 + tid;
#pragma unroll
  for (int i = 0; i < 4; ++i) aw[tid + i * 256] = attn[b * Sq + tid + i * 256];
  __syncthreads();
  const float* ep = enc + (size_t)b * Sq * Hq + h;
  float acc = 0.f;
#pragma unroll 8
  for (int s = 0; s < Sq; ++s) acc = fmaf(aw[s], ep[(size_t)s * Hq], acc);
  ctx[b * Hq + h] = acc;
}

// ---------------------------------------------------------------------------
extern "C" void kernel_launch(void* const* d_in, const int* in_sizes, int n_in,
                              void* d_out, int out_size, void* d_ws, size_t ws_size,
                              hipStream_t stream) {
  const float* dh  = (const float*)d_in[0];   // [B,H]
  const float* enc = (const float*)d_in[1];   // [B,S,H]
  const float* Wq  = (const float*)d_in[2];   // [H,H]
  const float* Wk  = (const float*)d_in[3];   // [H,H]
  const float* We  = (const float*)d_in[4];   // [H,1]

  float* ctx  = (float*)d_out;                // output 0: context [B,H]
  float* attn = (float*)d_out + Bq * Hq;      // output 1: attention [B,S,1]

  char* ws = (char*)d_ws;
  unsigned short* wkT = (unsigned short*)ws;                        // 2 MB bf16
  float* query  = (float*)(ws + (size_t)2 * 1024 * 1024);           // 128 KB
  float* energy = (float*)(ws + (size_t)2 * 1024 * 1024 + 131072);  // 128 KB

  wk_transpose_bf16<<<dim3(32, 32), dim3(32, 8), 0, stream>>>(Wk, wkT);
  query_gemm<<<dim3((Bq * Hq) / 256), dim3(256), 0, stream>>>(dh, Wq, query);
  keys_energy_wmma<<<dim3((Bq * Sq) / 64), dim3(256),
                     64 * 1040 * sizeof(unsigned short), stream>>>(
      enc, wkT, query, We, energy);
  softmax_s<<<dim3(Bq), dim3(256), 0, stream>>>(energy, attn);
  context_gemv<<<dim3(Hq / 256, Bq), dim3(256), 0, stream>>>(enc, attn, ctx);
}
